// GA_layer_33509334843923
// MI455X (gfx1250) — matrix operand order repaired and checked by reading, not compiled, not run on hardware
//
#include <hip/hip_runtime.h>
#include <hip/hip_bf16.h>

// ---------------------------------------------------------------------------
// PointNet++ set-abstraction layer for MI455X (gfx1250, wave32, WMMA).
//
//  1) fps_kernel   : farthest point sampling (B=16 blocks), writes new_xyz.
//  2) knn_kernel   : 32-NN per query (16384 blocks, LDS argmin extraction).
//  3) feat_kernel  : gather + [g-c, c] concat -> feat (f16, 524288 x 128).
//  4) gram_kernel  : Gram(feat) 128x128 + colsums; async global->LDS staging
//                    (global_load_async_to_lds_b128), LDS transpose, WMMA
//                    with all-vector (b128) LDS fragment reads.
//  5) bn_params    : analytic BN mean/var from Gram -> per-channel scale/shift.
//  6) gemm1_kernel : x1 = silu(bn(feat @ W1_0^T + b)) + feat (WMMA, b128 loads).
//  7) gram_kernel  : Gram(x1).
//  8) bn_params x2 : BN params for Ws_1 and W1_1.
//  9) gemm2_kernel : fused dual-GEMM + BN + silu + residual + max over 32
//                    samples + transposed store to d_out. (WMMA)
// ---------------------------------------------------------------------------

typedef __attribute__((ext_vector_type(16))) _Float16 v16h;
typedef __attribute__((ext_vector_type(8)))  _Float16 v8h;
typedef __attribute__((ext_vector_type(8)))  float    v8f;

#define NPTS   4096
#define NPOINT 1024
#define NSAMP  32
#define BATCH  16
#define CIN    128
#define TSTR   40        // padded column stride (halfs) of transposed LDS tile
#define ROWS   ((size_t)BATCH * NPOINT * NSAMP)   // 524288

__device__ __forceinline__ v8f wmma_f16(v16h a, v16h b, v8f c) {
  return __builtin_amdgcn_wmma_f32_16x16x32_f16(false, a, false, b,
                                                (short)0, c, false, false);
}

__device__ __forceinline__ v16h cat8(v8h lo, v8h hi) {
  return __builtin_shufflevector(lo, hi, 0, 1, 2, 3, 4, 5, 6, 7,
                                 8, 9, 10, 11, 12, 13, 14, 15);
}

// A fragment: 16x32 f16 tile A[m][k] from row-major src (leading dim ld halfs).
// Per-lane data = two contiguous 16B segments -> two b128 loads.
__device__ __forceinline__ v16h load_a_rm(const _Float16* src, int ld) {
  int lane = threadIdx.x & 31;
  int m  = lane & 15;
  int kb = (lane >> 4) << 4;          // 0 or 16
  const _Float16* p = src + m * ld + kb;
  return cat8(*(const v8h*)(p), *(const v8h*)(p + 8));
}

// B fragment B[k][n] = W[n][k] from row-major W. Per-lane data = 16
// contiguous halfs (32B, 32B-aligned) -> single v16h (2x b128) load.
__device__ __forceinline__ v16h load_b_wt(const _Float16* w, int ld) {
  int lane = threadIdx.x & 31;
  int n  = lane & 15;
  int kb = (lane >> 4) << 4;          // 0 or 16
  return *(const v16h*)(w + n * ld + kb);
}

// Gram fragment loads from the column-major (transposed, padded) LDS tile:
// tileT[c * TSTR + r] = X[r][c].  All segments 16B-aligned -> ds_load_b128.
__device__ __forceinline__ v16h ldsT_a(const _Float16* tileT, int colbase) {
  int lane = threadIdx.x & 31;
  int m  = colbase + (lane & 15);     // Gram row = column of X
  int kb = (lane >> 4) << 4;          // 0 or 16
  const _Float16* p = tileT + m * TSTR + kb;
  return cat8(*(const v8h*)(p), *(const v8h*)(p + 8));
}
__device__ __forceinline__ v16h ldsT_b(const _Float16* tileT, int colbase) {
  int lane = threadIdx.x & 31;
  int n  = colbase + (lane & 15);
  int kb = (lane >> 4) << 4;          // 0 or 16
  const _Float16* p = tileT + n * TSTR + kb;
  return cat8(*(const v8h*)(p), *(const v8h*)(p + 8));
}

// ------------------------------ 1) FPS -------------------------------------
__global__ void fps_kernel(const float* __restrict__ xyz,
                           int* __restrict__ fps_idx,
                           float* __restrict__ new_xyz) {
  const int b = blockIdx.x;
  const int t = threadIdx.x;                 // 256 threads
  const float* X = xyz + (size_t)b * NPTS * 3;

  float px[16], py[16], pz[16], dist[16];
#pragma unroll
  for (int i = 0; i < 16; ++i) {
    int p = t * 16 + i;
    px[i] = X[p * 3 + 0]; py[i] = X[p * 3 + 1]; pz[i] = X[p * 3 + 2];
    dist[i] = 1e10f;
  }

  __shared__ float sval[256];
  __shared__ int   sidx[256];
  __shared__ float cen[3];
  __shared__ int   cur;
  if (t == 0) cur = 0;
  __syncthreads();

  for (int it = 0; it < NPOINT; ++it) {
    if (t == 0) {
      int far = cur;
      fps_idx[b * NPOINT + it] = far;
      float cx = X[far * 3 + 0], cy = X[far * 3 + 1], cz = X[far * 3 + 2];
      cen[0] = cx; cen[1] = cy; cen[2] = cz;
      new_xyz[(size_t)(b * NPOINT + it) * 3 + 0] = cx;
      new_xyz[(size_t)(b * NPOINT + it) * 3 + 1] = cy;
      new_xyz[(size_t)(b * NPOINT + it) * 3 + 2] = cz;
    }
    __syncthreads();
    float cx = cen[0], cy = cen[1], cz = cen[2];
    float best = -1.0f; int bi = t * 16;
#pragma unroll
    for (int i = 0; i < 16; ++i) {
      float dx = px[i] - cx, dy = py[i] - cy, dz = pz[i] - cz;
      float d = dx * dx + dy * dy + dz * dz;
      dist[i] = fminf(dist[i], d);
      if (dist[i] > best) { best = dist[i]; bi = t * 16 + i; }
    }
    sval[t] = best; sidx[t] = bi;
    __syncthreads();
    for (int off = 128; off > 0; off >>= 1) {
      if (t < off && sval[t + off] > sval[t]) {
        sval[t] = sval[t + off]; sidx[t] = sidx[t + off];
      }
      __syncthreads();
    }
    if (t == 0) cur = sidx[0];
    __syncthreads();
  }
}

// ------------------------------ 2) KNN -------------------------------------
__global__ void knn_kernel(const float* __restrict__ xyz,
                           const int* __restrict__ fps_idx,
                           int* __restrict__ knn_idx) {
  const int blk = blockIdx.x;          // b*1024 + s
  const int b = blk >> 10;
  const int t = threadIdx.x;           // 256 threads
  const float* X = xyz + (size_t)b * NPTS * 3;
  const int q = fps_idx[blk];
  const float qx = X[q * 3 + 0], qy = X[q * 3 + 1], qz = X[q * 3 + 2];

  __shared__ float d[NPTS];
  __shared__ float sval[256];
  __shared__ int   sidx[256];

#pragma unroll
  for (int i = 0; i < 16; ++i) {
    int p = t + i * 256;
    float dx = X[p * 3 + 0] - qx;
    float dy = X[p * 3 + 1] - qy;
    float dz = X[p * 3 + 2] - qz;
    d[p] = dx * dx + dy * dy + dz * dz;
  }
  __syncthreads();

  for (int kk = 0; kk < NSAMP; ++kk) {
    float best = 1e30f; int bi = t;
#pragma unroll
    for (int i = 0; i < 16; ++i) {
      int p = t + i * 256;
      if (d[p] < best) { best = d[p]; bi = p; }
    }
    sval[t] = best; sidx[t] = bi;
    __syncthreads();
    for (int off = 128; off > 0; off >>= 1) {
      if (t < off && sval[t + off] < sval[t]) {
        sval[t] = sval[t + off]; sidx[t] = sidx[t + off];
      }
      __syncthreads();
    }
    if (t == 0) {
      knn_idx[(size_t)blk * NSAMP + kk] = sidx[0];
      d[sidx[0]] = 1e30f;
    }
    __syncthreads();
  }
}

// --------------------- 3) feature build (f16) ------------------------------
__global__ void feat_kernel(const float* __restrict__ points,
                            const int* __restrict__ fps_idx,
                            const int* __restrict__ knn_idx,
                            _Float16* __restrict__ feat) {
  const int blk = blockIdx.x;          // b*1024 + s
  const int b = blk >> 10;
  const int t = threadIdx.x;           // 128 threads
  const float* P = points + (size_t)b * NPTS * 64;

  __shared__ float c[64];
  int ci = fps_idx[blk];
  if (t < 64) c[t] = P[(size_t)ci * 64 + t];
  __syncthreads();

  _Float16* F = feat + (size_t)blk * NSAMP * CIN;
  for (int k = 0; k < NSAMP; ++k) {
    int gi = knn_idx[(size_t)blk * NSAMP + k];
    float v = (t < 64) ? (P[(size_t)gi * 64 + t] - c[t]) : c[t - 64];
    F[k * CIN + t] = (_Float16)v;
  }
}

// --------------------- weight f32 -> f16 conversion ------------------------
__global__ void cvt_w_kernel(const float* __restrict__ w10,
                             const float* __restrict__ ws1,
                             const float* __restrict__ w11,
                             _Float16* __restrict__ out) {
  int i = blockIdx.x * 256 + threadIdx.x;
  if (i < 128 * 128) out[i] = (_Float16)w10[i];
  if (i < 256 * 128) {
    out[128 * 128 + i]             = (_Float16)ws1[i];
    out[128 * 128 + 256 * 128 + i] = (_Float16)w11[i];
  }
}

// --------------- 4/7) Gram matrix (128x128) + column sums ------------------
// X: [ROWS][128] f16. 256 blocks x 256 threads (8 waves); chunk = 2048 rows.
// Per 32-row slab: async global->LDS row staging (ASYNCcnt), LDS transpose
// into padded column-major tile, then all-vector ds_load_b128 WMMA fragments.
__global__ void gram_kernel(const _Float16* __restrict__ X,
                            float* __restrict__ G,
                            float* __restrict__ csum) {
  __shared__ __align__(32) _Float16 rowbuf[32 * CIN];     //  8 KB, row-major
  __shared__ __align__(32) _Float16 tileT[CIN * TSTR];    // 10 KB, col-major
  const size_t rowbase = (size_t)blockIdx.x * 2048;
  const int t = threadIdx.x, wave = t >> 5, lane = t & 31;
  const int ti = wave;                                    // row tile of G
  const int tr = t >> 3;                                  // transpose: row
  const int tc = (t & 7) * 16;                            // transpose: col base

  v8f zero = {};
  v8f acc[8];
#pragma unroll
  for (int j = 0; j < 8; ++j) acc[j] = zero;
  float cs = 0.f;

  unsigned ldsoff = (unsigned)(size_t)&rowbuf[t * 16];    // 32B per thread

  for (int ks = 0; ks < 64; ++ks) {
    unsigned long long ga =
        (unsigned long long)(X + (rowbase + (size_t)ks * 32) * CIN + t * 16);
    // async copy 32B (two b128) global -> LDS; offset applies to both sides
    asm volatile("global_load_async_to_lds_b128 %0, %1, off\n\t"
                 "global_load_async_to_lds_b128 %0, %1, off offset:16"
                 :: "v"(ldsoff), "v"(ga) : "memory");
    asm volatile("s_wait_asynccnt 0" ::: "memory");
    __syncthreads();

    // LDS transpose: vector read from rowbuf, scatter into tileT
    {
      v8h s0 = *(const v8h*)(rowbuf + tr * CIN + tc);
      v8h s1 = *(const v8h*)(rowbuf + tr * CIN + tc + 8);
#pragma unroll
      for (int j = 0; j < 8; ++j) {
        tileT[(tc + j) * TSTR + tr]     = s0[j];
        tileT[(tc + 8 + j) * TSTR + tr] = s1[j];
      }
    }
    __syncthreads();

    v16h a = ldsT_a(tileT, ti * 16);
#pragma unroll
    for (int tj = 0; tj < 8; ++tj) {
      v16h bf = ldsT_b(tileT, tj * 16);
      acc[tj] = wmma_f16(a, bf, acc[tj]);
    }
    if (t < CIN) {                       // column sums: contiguous in tileT
      float s = 0.f;
#pragma unroll
      for (int r = 0; r < 32; ++r) s += (float)tileT[t * TSTR + r];
      cs += s;
    }
    __syncthreads();
  }
#pragma unroll
  for (int tj = 0; tj < 8; ++tj) {
#pragma unroll
    for (int r = 0; r < 8; ++r) {
      int row = ti * 16 + r + ((lane >> 4) << 3);
      int col = tj * 16 + (lane & 15);
      atomicAdd(&G[row * CIN + col], acc[tj][r]);
    }
  }
  if (t < CIN) atomicAdd(&csum[t], cs);
}

// ---------- 5/8) analytic BN params from Gram: scale/shift per channel -----
__global__ void bn_params_kernel(int nch,
                                 const float* __restrict__ W,     // [nch][128]
                                 const float* __restrict__ bias,
                                 const float* __restrict__ g,
                                 const float* __restrict__ bt,
                                 const float* __restrict__ G,
                                 const float* __restrict__ s,
                                 float invM,
                                 float* __restrict__ scale,
                                 float* __restrict__ shiftT) {
  int o = blockIdx.x * 64 + threadIdx.x;
  if (o >= nch) return;
  const float* w = W + (size_t)o * CIN;
  float ws = 0.f;
  for (int c = 0; c < CIN; ++c) ws += w[c] * s[c];
  float m = ws * invM + bias[o];
  float q = 0.f;
  for (int i = 0; i < CIN; ++i) {
    const float* Gi = G + i * CIN;
    float a = 0.f;
    for (int j = 0; j < CIN; ++j) a += Gi[j] * w[j];
    q += w[i] * a;
  }
  float bo = bias[o];
  float ey2 = q * invM + 2.f * bo * (m - bo) + bo * bo;
  float var = ey2 - m * m;
  float sc = g[o] * rsqrtf(var + 1e-5f);
  scale[o]  = sc;
  shiftT[o] = bt[o] + sc * (bo - m);   // applied to raw dot (bias folded in)
}

// ---- 6) GEMM1: x1 = silu(scale*(feat@W^T)+shift) + feat, f16 out ----------
// 2048 blocks x 256 threads (8 waves); wave owns 32 rows, full 128 cols.
__global__ void gemm1_kernel(const _Float16* __restrict__ feat,
                             const _Float16* __restrict__ Wh,   // [128][128]
                             const float* __restrict__ scale,
                             const float* __restrict__ shift,
                             _Float16* __restrict__ x1) {
  const int t = threadIdx.x, wave = t >> 5, lane = t & 31;
  const size_t rowbase = (size_t)blockIdx.x * 256 + (size_t)wave * 32;
  __builtin_prefetch(Wh, 0, 1);                 // global_prefetch_b8

  v8f zero = {};
  v8f acc[2][8];
#pragma unroll
  for (int m = 0; m < 2; ++m)
#pragma unroll
    for (int n = 0; n < 8; ++n) acc[m][n] = zero;

#pragma unroll
  for (int ks = 0; ks < 4; ++ks) {
    v16h a0 = load_a_rm(feat + rowbase * CIN + ks * 32, CIN);
    v16h a1 = load_a_rm(feat + (rowbase + 16) * CIN + ks * 32, CIN);
#pragma unroll
    for (int nt = 0; nt < 8; ++nt) {
      v16h bf = load_b_wt(Wh + (size_t)(nt * 16) * CIN + ks * 32, CIN);
      acc[0][nt] = wmma_f16(a0, bf, acc[0][nt]);
      acc[1][nt] = wmma_f16(a1, bf, acc[1][nt]);
    }
  }
#pragma unroll
  for (int mt = 0; mt < 2; ++mt) {
#pragma unroll
    for (int nt = 0; nt < 8; ++nt) {
      int col = nt * 16 + (lane & 15);
      float sc = scale[col], sh = shift[col];
#pragma unroll
      for (int r = 0; r < 8; ++r) {
        size_t row = rowbase + mt * 16 + r + ((lane >> 4) << 3);
        float tt = sc * acc[mt][nt][r] + sh;
        float sig = 1.f / (1.f + __expf(-tt));
        float v = tt * sig + (float)feat[row * CIN + col];
        x1[row * CIN + col] = (_Float16)v;
      }
    }
  }
}

// ---- 9) GEMM2 fused: res/x2 + silu + residual + max over 32 + transpose ---
// 16384 blocks (one (b,s) group of 32 rows) x 256 threads; wave w owns
// output-channel range [w*32, w*32+32) of BOTH weight sets.
__global__ void gemm2_kernel(const _Float16* __restrict__ x1,
                             const _Float16* __restrict__ WhS,  // [256][128]
                             const _Float16* __restrict__ Wh1,  // [256][128]
                             const float* __restrict__ scS,
                             const float* __restrict__ shS,
                             const float* __restrict__ sc1,
                             const float* __restrict__ sh1,
                             float* __restrict__ outT) {        // [B][256][1024]
  const int blk = blockIdx.x;
  const int b = blk >> 10, s = blk & 1023;
  const int t = threadIdx.x, wave = t >> 5, lane = t & 31;
  const size_t rowbase = (size_t)blk * 32;
  const int ntb = wave * 32;
  __builtin_prefetch(WhS, 0, 1);
  __builtin_prefetch(Wh1, 0, 1);

  v8f zero = {};
  v8f accS[2][2], acc1[2][2];
#pragma unroll
  for (int m = 0; m < 2; ++m)
#pragma unroll
    for (int n = 0; n < 2; ++n) { accS[m][n] = zero; acc1[m][n] = zero; }

#pragma unroll
  for (int ks = 0; ks < 4; ++ks) {
    v16h a0 = load_a_rm(x1 + rowbase * CIN + ks * 32, CIN);
    v16h a1 = load_a_rm(x1 + (rowbase + 16) * CIN + ks * 32, CIN);
#pragma unroll
    for (int nt = 0; nt < 2; ++nt) {
      v16h bS = load_b_wt(WhS + (size_t)(ntb + nt * 16) * CIN + ks * 32, CIN);
      v16h b1 = load_b_wt(Wh1 + (size_t)(ntb + nt * 16) * CIN + ks * 32, CIN);
      accS[0][nt] = wmma_f16(a0, bS, accS[0][nt]);
      accS[1][nt] = wmma_f16(a1, bS, accS[1][nt]);
      acc1[0][nt] = wmma_f16(a0, b1, acc1[0][nt]);
      acc1[1][nt] = wmma_f16(a1, b1, acc1[1][nt]);
    }
  }
#pragma unroll
  for (int nt = 0; nt < 2; ++nt) {
    int col = ntb + nt * 16 + (lane & 15);
    float sS = scS[col], hS = shS[col];
    float s1 = sc1[col], h1 = sh1[col];
    float mx = -1e30f;
#pragma unroll
    for (int mt = 0; mt < 2; ++mt) {
#pragma unroll
      for (int r = 0; r < 8; ++r) {
        float res = sS * accS[mt][nt][r] + hS;
        float tt  = s1 * acc1[mt][nt][r] + h1;
        float sig = 1.f / (1.f + __expf(-tt));
        mx = fmaxf(mx, tt * sig + res);
      }
    }
    mx = fmaxf(mx, __shfl_xor(mx, 16, 32));   // merge M halves (wave32)
    if (lane < 16)
      outT[((size_t)b * 256 + col) * NPOINT + s] = mx;
  }
}

// ---------------------------------------------------------------------------
extern "C" void kernel_launch(void* const* d_in, const int* in_sizes, int n_in,
                              void* d_out, int out_size, void* d_ws, size_t ws_size,
                              hipStream_t stream) {
  const float* xyz   = (const float*)d_in[0];
  const float* points= (const float*)d_in[1];
  const float* W1_0  = (const float*)d_in[2];
  const float* b1_0  = (const float*)d_in[3];
  const float* g1_0  = (const float*)d_in[4];
  const float* bt1_0 = (const float*)d_in[5];
  const float* W1_1  = (const float*)d_in[6];
  const float* b1_1  = (const float*)d_in[7];
  const float* g1_1  = (const float*)d_in[8];
  const float* bt1_1 = (const float*)d_in[9];
  const float* Ws_1  = (const float*)d_in[10];
  const float* bs_1  = (const float*)d_in[11];
  const float* gs_1  = (const float*)d_in[12];
  const float* bts_1 = (const float*)d_in[13];

  char* ws = (char*)d_ws;
  // workspace layout (bytes, all 256-aligned); total ~259 MB
  int*      fps_idx = (int*)(ws + 0);                         //   64 KB
  int*      knn_idx = (int*)(ws + 65536);                     //    2 MB
  _Float16* wh      = (_Float16*)(ws + 2162688);              //  160 KB (f16 weights)
  float*    G0      = (float*)(ws + 2326528);                 //   64 KB
  float*    s0      = (float*)(ws + 2392064);                 //  512 B
  float*    G1      = (float*)(ws + 2392576);                 //   64 KB
  float*    s1      = (float*)(ws + 2458112);                 //  512 B
  float*    sc1     = (float*)(ws + 2458624);
  float*    sh1     = (float*)(ws + 2459136);
  float*    scS     = (float*)(ws + 2459648);
  float*    shS     = (float*)(ws + 2460672);
  float*    sc1b    = (float*)(ws + 2461696);
  float*    sh1b    = (float*)(ws + 2462720);
  _Float16* feat    = (_Float16*)(ws + 2463744);              //  128 MB
  _Float16* x1      = (_Float16*)(ws + 2463744 + 134217728);  //  128 MB

  _Float16* whS = wh + 128 * 128;
  _Float16* wh1 = wh + 128 * 128 + 256 * 128;

  float* out      = (float*)d_out;
  float* new_xyz  = out;                       // B*1024*3 = 49152 floats
  float* outT     = out + (size_t)BATCH * NPOINT * 3;

  const float invM = 1.0f / (float)ROWS;

  cvt_w_kernel<<<128, 256, 0, stream>>>(W1_0, Ws_1, W1_1, wh);
  fps_kernel<<<BATCH, 256, 0, stream>>>(xyz, fps_idx, new_xyz);
  knn_kernel<<<BATCH * NPOINT, 256, 0, stream>>>(xyz, fps_idx, knn_idx);
  feat_kernel<<<BATCH * NPOINT, 128, 0, stream>>>(points, fps_idx, knn_idx, feat);

  // zero Gram/colsum accumulators (G0,s0,G1,s1 are contiguous: 132096 bytes)
  hipMemsetAsync(G0, 0, 132096, stream);

  gram_kernel<<<256, 256, 0, stream>>>(feat, G0, s0);
  bn_params_kernel<<<2, 64, 0, stream>>>(128, W1_0, b1_0, g1_0, bt1_0,
                                         G0, s0, invM, sc1, sh1);
  gemm1_kernel<<<(int)(ROWS / 256), 256, 0, stream>>>(feat, wh, sc1, sh1, x1);

  gram_kernel<<<256, 256, 0, stream>>>(x1, G1, s1);
  bn_params_kernel<<<4, 64, 0, stream>>>(256, Ws_1, bs_1, gs_1, bts_1,
                                         G1, s1, invM, scS, shS);
  bn_params_kernel<<<4, 64, 0, stream>>>(256, W1_1, b1_1, g1_1, bt1_1,
                                         G1, s1, invM, sc1b, sh1b);

  gemm2_kernel<<<BATCH * NPOINT, 256, 0, stream>>>(x1, whS, wh1,
                                                   scS, shS, sc1b, sh1b, outT);
}
